// PAM_moudle_44908178046992
// MI455X (gfx1250) — matrix operand order repaired and checked by reading, not compiled, not run on hardware
//
#include <hip/hip_runtime.h>
#include <stdint.h>

typedef _Float16 half_t;
typedef __attribute__((ext_vector_type(16))) _Float16 v16h;
typedef __attribute__((ext_vector_type(8)))  _Float16 v8h;
typedef __attribute__((ext_vector_type(8)))  float    v8f;

#define Bb 2
#define Cc 64
#define CQd 8
#define Nn 9216          // 96*96
#define NT (Nn / 16)     // 576 i-tiles per batch
#define WAVES 4
#define NCH (Nn / 32)    // 288 j-chunks

// ---------------- Phase 1: 1x1-conv projections (q, k^T, v) to f16 ----------------
__global__ __launch_bounds__(256) void proj_kernel(
    const float* __restrict__ X,
    const float* __restrict__ Wq, const float* __restrict__ bq,
    const float* __restrict__ Wk, const float* __restrict__ bk,
    const float* __restrict__ Wv, const float* __restrict__ bv,
    half_t* __restrict__ Qh, half_t* __restrict__ KTh, half_t* __restrict__ Vh)
{
    int p = blockIdx.x * blockDim.x + threadIdx.x;
    if (p >= Bb * Nn) return;
    int b = p / Nn, n = p % Nn;

    const float* Xb = X + (size_t)b * Cc * Nn + n;
    float x[Cc];
#pragma unroll
    for (int c = 0; c < Cc; ++c) x[c] = Xb[(size_t)c * Nn];

    half_t* qrow = Qh  + ((size_t)b * Nn + n) * CQd;
    half_t* krow = KTh + ((size_t)b * Nn + n) * CQd;
#pragma unroll
    for (int qi = 0; qi < CQd; ++qi) {
        float aq = bq[qi], ak = bk[qi];
        const float* wq = Wq + qi * Cc;
        const float* wk = Wk + qi * Cc;
#pragma unroll
        for (int c = 0; c < Cc; ++c) {
            aq = fmaf(wq[c], x[c], aq);
            ak = fmaf(wk[c], x[c], ak);
        }
        qrow[qi] = (half_t)aq;
        krow[qi] = (half_t)ak;
    }
    for (int co = 0; co < Cc; ++co) {
        float av = bv[co];
        const float* wv = Wv + co * Cc;
#pragma unroll
        for (int c = 0; c < Cc; ++c) av = fmaf(wv[c], x[c], av);
        Vh[((size_t)b * Cc + co) * Nn + n] = (half_t)av;
    }
}

// ---------------- Phase 2: fused flash-style attention + gamma*out + X ----------------
// One wave per 16-row i-tile; 4 waves/block share async-staged V tiles in LDS.
// WMMA operand layouts per CDNA5 ISA 7.12.2 (A 16x32 f16, B 32x16 f16, C/D 16x16 f32).
__global__ __launch_bounds__(32 * WAVES) void attn_kernel(
    const float* __restrict__ X,
    const half_t* __restrict__ Qh, const half_t* __restrict__ KTh,
    const half_t* __restrict__ Vh, const float* __restrict__ gamma,
    float* __restrict__ out)
{
    __shared__ __align__(64) _Float16 vbuf[2][Cc][32];      // ping-pong async V tile (2 x 4KB)
    __shared__ __align__(64) _Float16 pbuf[WAVES][16][32];  // P tile staging per wave
    __shared__ float sbuf[WAVES][16];                       // row sums per wave

    const int lane   = threadIdx.x & 31;
    const int w      = threadIdx.x >> 5;
    const int tile   = blockIdx.x * WAVES + w;
    const int b      = tile / NT;                // uniform per block (576 % WAVES == 0)
    const int i0     = (tile % NT) * 16;
    const int half16 = lane >> 4;                // 0: lanes 0-15, 1: lanes 16-31
    const int col0   = lane & 15;

    const half_t* Qb = Qh  + (size_t)b * Nn * CQd;
    const half_t* Kb = KTh + (size_t)b * Nn * CQd;
    // staging base: uniform across the block -> lives in an SGPR pair for async loads
    const unsigned long long VbAddr =
        (unsigned long long)(uintptr_t)(Vh + (size_t)b * Cc * Nn);

    // per-thread staging slice: 2 x b128 segments of the 64x32 f16 V tile (4KB / 128 thr)
    const int s0    = (int)threadIdx.x * 2;
    const int row0  = s0 >> 2, seg0 = s0 & 3;
    const int row1  = (s0 + 1) >> 2, seg1 = (s0 + 1) & 3;
    unsigned gof0 = (unsigned)((row0 * Nn + seg0 * 8) * 2);
    unsigned gof1 = (unsigned)((row1 * Nn + seg1 * 8) * 2);

#define STAGE_V(J0, BUF)                                                              \
    do {                                                                              \
        unsigned l0 = (unsigned)(uintptr_t)&vbuf[(BUF)][row0][seg0 * 8];              \
        unsigned l1 = (unsigned)(uintptr_t)&vbuf[(BUF)][row1][seg1 * 8];              \
        unsigned g0 = gof0 + (unsigned)((J0) * 2);                                    \
        unsigned g1 = gof1 + (unsigned)((J0) * 2);                                    \
        asm volatile("global_load_async_to_lds_b128 %0, %1, %2"                       \
                     :: "v"(l0), "v"(g0), "s"(VbAddr) : "memory");                    \
        asm volatile("global_load_async_to_lds_b128 %0, %1, %2"                       \
                     :: "v"(l1), "v"(g1), "s"(VbAddr) : "memory");                    \
    } while (0)

    // Q A-operand: row i0+lane (lanes<16), K = 0..7 real, rest zero-pad.
    v16h aq;
#pragma unroll
    for (int e = 0; e < 16; ++e) aq[e] = (_Float16)0.0f;
    if (lane < 16) {
        v8h t = *(const v8h*)(Qb + (size_t)(i0 + lane) * CQd);
#pragma unroll
        for (int e = 0; e < 8; ++e) aq[e] = t[e];
    }

    v8f z8;
#pragma unroll
    for (int r = 0; r < 8; ++r) z8[r] = 0.0f;

    v8f acc[4];
#pragma unroll
    for (int cc = 0; cc < 4; ++cc) acc[cc] = z8;
    float rsum[8];
#pragma unroll
    for (int r = 0; r < 8; ++r) rsum[r] = 0.0f;

    STAGE_V(0, 0);                    // prefetch chunk 0 (ASYNCcnt = 2)

    for (int ch = 0; ch < NCH; ++ch) {
        const int j0  = ch * 32;
        const int buf = ch & 1;

        // issue next chunk's DMA, then wait only for the current chunk (in-order done)
        if (ch + 1 < NCH) {
            STAGE_V(j0 + 32, buf ^ 1);
            asm volatile("s_wait_asynccnt 0x2" ::: "memory");
        } else {
            asm volatile("s_wait_asynccnt 0x0" ::: "memory");
        }
        __syncthreads();              // all waves' staged slices visible

        // ---- S = Q * K^T (two 16x16 tiles covering cols j0..j0+31) ----
        v16h bk0, bk1;
#pragma unroll
        for (int e = 0; e < 16; ++e) { bk0[e] = (_Float16)0.0f; bk1[e] = (_Float16)0.0f; }
        if (lane < 16) {
            v8h t0 = *(const v8h*)(Kb + (size_t)(j0 + lane) * CQd);
            v8h t1 = *(const v8h*)(Kb + (size_t)(j0 + 16 + lane) * CQd);
#pragma unroll
            for (int e = 0; e < 8; ++e) { bk0[e] = t0[e]; bk1[e] = t1[e]; }
        }
        if (j0 + 32 < Nn)
            __builtin_prefetch(Kb + (size_t)(j0 + 32 + lane) * CQd, 0, 1);

        v8f s0v = __builtin_amdgcn_wmma_f32_16x16x32_f16(false, aq, false, bk0,
                                                         (short)0, z8, false, false);
        v8f s1v = __builtin_amdgcn_wmma_f32_16x16x32_f16(false, aq, false, bk1,
                                                         (short)0, z8, false, false);

        // ---- P = exp(S) (no max-shift needed: |S| <~ 12), row-sum, stage to LDS ----
#pragma unroll
        for (int r = 0; r < 8; ++r) {
            float p0 = __expf(s0v[r]);
            float p1 = __expf(s1v[r]);
            rsum[r] += p0 + p1;
            int rr = r + half16 * 8;               // absolute i-local row
            pbuf[w][rr][col0]      = (half_t)p0;   // col j-local = col0
            pbuf[w][rr][col0 + 16] = (half_t)p1;   // col j-local = col0+16
        }
        asm volatile("s_wait_dscnt 0x0" ::: "memory");
        __builtin_amdgcn_wave_barrier();

        // ---- P^T B-operand: lane n<16 -> row n cols 0..15; lane n>=16 -> row n-16 cols 16..31
        v16h bp = *(const v16h*)(&pbuf[w][col0][half16 * 16]);

        // ---- O^T += V * P^T over all 64 channels; A-tiles read from LDS-staged V ----
#pragma unroll
        for (int cc = 0; cc < 4; ++cc) {
            const int crow = cc * 16 + col0;
            v8h lo = *(const v8h*)(&vbuf[buf][crow][half16 * 16]);      // K 0..7 / 16..23
            v8h hi = *(const v8h*)(&vbuf[buf][crow][half16 * 16 + 8]);  // K 8..15 / 24..31
            v16h av;
#pragma unroll
            for (int e = 0; e < 8; ++e) { av[e] = lo[e]; av[e + 8] = hi[e]; }
            acc[cc] = __builtin_amdgcn_wmma_f32_16x16x32_f16(false, av, false, bp,
                                                             (short)0, acc[cc], false, false);
        }
        __syncthreads();              // vbuf[buf] free for the DMA issued next iteration
    }

    // ---- finalize row sums: reduce within each 16-lane half, publish via LDS ----
#pragma unroll
    for (int r = 0; r < 8; ++r) {
#pragma unroll
        for (int off = 8; off >= 1; off >>= 1)
            rsum[r] += __shfl_xor(rsum[r], off, 32);
    }
    if (col0 == 0) {
#pragma unroll
        for (int r = 0; r < 8; ++r) sbuf[w][half16 * 8 + r] = rsum[r];
    }
    asm volatile("s_wait_dscnt 0x0" ::: "memory");
    __builtin_amdgcn_wave_barrier();

    const float inv = 1.0f / sbuf[w][col0];    // this lane's i-column rowsum
    const float g   = gamma[0];
    const float* Xb2  = X   + (size_t)b * Cc * Nn;
    float*       outb = out + (size_t)b * Cc * Nn;

#pragma unroll
    for (int cc = 0; cc < 4; ++cc) {
#pragma unroll
        for (int r = 0; r < 8; ++r) {
            int c = cc * 16 + r + half16 * 8;          // D-tile row M -> channel
            size_t idx = (size_t)c * Nn + i0 + col0;   // D-tile col N -> pixel
            outb[idx] = fmaf(g, acc[cc][r] * inv, Xb2[idx]);
        }
    }
#undef STAGE_V
}

extern "C" void kernel_launch(void* const* d_in, const int* in_sizes, int n_in,
                              void* d_out, int out_size, void* d_ws, size_t ws_size,
                              hipStream_t stream) {
    const float* X     = (const float*)d_in[0];
    const float* Wq    = (const float*)d_in[1];
    const float* bq    = (const float*)d_in[2];
    const float* Wk    = (const float*)d_in[3];
    const float* bk    = (const float*)d_in[4];
    const float* Wv    = (const float*)d_in[5];
    const float* bv    = (const float*)d_in[6];
    const float* gamma = (const float*)d_in[7];
    float* out = (float*)d_out;

    // workspace layout (all 16B-aligned): Q (B,N,8) f16 | K^T (B,N,8) f16 | V (B,C,N) f16
    const size_t qBytes = (size_t)Bb * Nn * CQd * sizeof(half_t);   // 294912
    half_t* Qh  = (half_t*)d_ws;
    half_t* KTh = (half_t*)((char*)d_ws + qBytes);
    half_t* Vh  = (half_t*)((char*)d_ws + 2 * qBytes);

    proj_kernel<<<(Bb * Nn + 255) / 256, 256, 0, stream>>>(
        X, Wq, bq, Wk, bk, Wv, bv, Qh, KTh, Vh);

    attn_kernel<<<(Bb * NT) / WAVES, 32 * WAVES, 0, stream>>>(
        X, Qh, KTh, Vh, gamma, out);
}
